// MoEFSCIL_30554397343881
// MI455X (gfx1250) — matrix operand order repaired and checked by reading, not compiled, not run on hardware
//
#include <hip/hip_runtime.h>
#include <hip/hip_bf16.h>
#include <math.h>

#define DIM   512
#define E_    4
#define K_    4
#define R_    4
#define B_    32
#define H_    14
#define W_    14
#define L_    (H_*W_)      // 196
#define NPOS  (B_*L_)      // 6272
#define TOPK_ 2
#define AUXW_ 0.01f

typedef __attribute__((ext_vector_type(16))) __bf16         v16bf;
typedef __attribute__((ext_vector_type(8)))  __bf16         v8bf;
typedef __attribute__((ext_vector_type(8)))  float          v8f;
typedef __attribute__((ext_vector_type(8)))  unsigned short v8us;

// ---------------- helpers ----------------
__device__ __forceinline__ unsigned short f32_to_bf16_rne(float f) {
    unsigned int u = __float_as_uint(f);
    unsigned int r = u + 0x7fffu + ((u >> 16) & 1u);
    return (unsigned short)(r >> 16);
}

__device__ __forceinline__ float silu(float v) { return v / (1.f + __expf(-v)); }

// position map for scan direction k at step l -> flat h*W+w index (same map
// is used for gather on input and scatter on output — flips/transposes cancel)
__device__ __forceinline__ int scan_map(int k, int l) {
    int m = (k & 1) ? (L_ - 1 - l) : l;
    if (k >= 2) m = (m % H_) * W_ + m / H_;
    return m;
}

__device__ __forceinline__ v16bf ld_frag32(const unsigned short* p0, const unsigned short* p1) {
    v8us a = *(const v8us*)p0;
    v8us b = *(const v8us*)p1;
    return __builtin_shufflevector(__builtin_bit_cast(v8bf, a), __builtin_bit_cast(v8bf, b),
                                   0,1,2,3,4,5,6,7,8,9,10,11,12,13,14,15);
}

// ---------------- utility kernels ----------------
__global__ void zero_f32(float* __restrict__ p, int n) {
    int i = blockIdx.x * blockDim.x + threadIdx.x;
    if (i < n) p[i] = 0.f;
}

__global__ void cast_bf16(const float* __restrict__ in, unsigned short* __restrict__ out, int n) {
    int i = blockIdx.x * blockDim.x + threadIdx.x;
    if (i < n) out[i] = f32_to_bf16_rne(in[i]);
}

// ---------------- gating: logits, softmax, top-2, aux accumulators ----------------
__global__ void __launch_bounds__(256)
gating_kernel(const float* __restrict__ x, const float* __restrict__ q,
              const float* __restrict__ temp, float* __restrict__ comb,
              float* __restrict__ aux /* [0..3]=sum gate, [4..7]=top2 counts */) {
    int wave = threadIdx.x >> 5, lane = threadIdx.x & 31;
    int p = blockIdx.x * 8 + wave;
    if (p >= NPOS) return;
    const float* xr = x + (size_t)p * DIM;
    float a0 = 0.f, a1 = 0.f, a2 = 0.f, a3 = 0.f;
    for (int d = lane; d < DIM; d += 32) {
        float xv = xr[d];
        a0 += xv * q[d];
        a1 += xv * q[DIM + d];
        a2 += xv * q[2 * DIM + d];
        a3 += xv * q[3 * DIM + d];
    }
    #pragma unroll
    for (int m = 16; m >= 1; m >>= 1) {
        a0 += __shfl_xor(a0, m, 32);
        a1 += __shfl_xor(a1, m, 32);
        a2 += __shfl_xor(a2, m, 32);
        a3 += __shfl_xor(a3, m, 32);
    }
    if (lane == 0) {
        float t = temp[0];
        float lg[E_] = { a0 / t, a1 / t, a2 / t, a3 / t };
        float mx = fmaxf(fmaxf(lg[0], lg[1]), fmaxf(lg[2], lg[3]));
        float ex[E_], s = 0.f;
        #pragma unroll
        for (int e = 0; e < E_; e++) { ex[e] = __expf(lg[e] - mx); s += ex[e]; }
        float g[E_];
        #pragma unroll
        for (int e = 0; e < E_; e++) g[e] = ex[e] / s;
        int i1 = 0;
        #pragma unroll
        for (int e = 1; e < E_; e++) if (g[e] > g[i1]) i1 = e;   // first-index ties
        int i2 = -1;
        #pragma unroll
        for (int e = 0; e < E_; e++) {
            if (e == i1) continue;
            if (i2 < 0 || g[e] > g[i2]) i2 = e;
        }
        float wsum = g[i1] + g[i2];
        float cb[E_] = { 0.f, 0.f, 0.f, 0.f };
        cb[i1] = g[i1] / wsum;
        cb[i2] = g[i2] / wsum;
        #pragma unroll
        for (int e = 0; e < E_; e++) {
            comb[(size_t)p * E_ + e] = cb[e];
            atomicAdd(&aux[e], g[e]);
        }
        atomicAdd(&aux[E_ + i1], 1.0f);
        atomicAdd(&aux[E_ + i2], 1.0f);
    }
}

// ---------------- in_proj GEMM: C(6272x1024) = Xbf16(6272x512) * W^T via WMMA ----------------
// Register-blocked: each wave produces a 16x64 strip (4 N-tiles) so the A
// fragment is loaded once per 4 WMMAs. 392*16 strips, 4 waves/block.
__global__ void __launch_bounds__(128)
gemm_xz_bf16(const unsigned short* __restrict__ Xb,   // (NPOS, 512) bf16 bits, row-major
             const unsigned short* __restrict__ Wb,   // (1024, 512) bf16 bits, row-major (this expert)
             float* __restrict__ Cout) {              // (NPOS, 1024) fp32
    const int wave = threadIdx.x >> 5;
    const int lane = threadIdx.x & 31;
    const int stripsN = 1024 / 64;                    // 16
    int strip = blockIdx.x * 4 + wave;                // 392*16 strips total
    int tn = strip % stripsN, tm = strip / stripsN;
    const int half = lane >> 4;                       // lane group (0: lanes 0-15, 1: 16-31)
    const int mn   = lane & 15;                       // M row (A) / N col (B)
    const unsigned short* arow = Xb + (size_t)(tm * 16 + mn) * DIM;
    const unsigned short* brow = Wb + (size_t)(tn * 64 + mn) * DIM;   // +j*16 rows per sub-tile
    v8f acc0 = {}, acc1 = {}, acc2 = {}, acc3 = {};
    #pragma unroll 2
    for (int kk = 0; kk < DIM; kk += 32) {
        // A 16x32 bf16 fragment: lane<16 holds K 0..15, lane>=16 K 16..31
        v16bf A = ld_frag32(arow + kk + half * 16, arow + kk + half * 16 + 8);
        // B 32x16 fragments: lanes 0-15 K=0..15, lanes 16-31 K=16..31 along k of W row
        const unsigned short* bk = brow + kk + half * 16;
        v16bf B0 = ld_frag32(bk,                  bk + 8);
        v16bf B1 = ld_frag32(bk + 16 * DIM,       bk + 16 * DIM + 8);
        v16bf B2 = ld_frag32(bk + 32 * DIM,       bk + 32 * DIM + 8);
        v16bf B3 = ld_frag32(bk + 48 * DIM,       bk + 48 * DIM + 8);
        acc0 = __builtin_amdgcn_wmma_f32_16x16x32_bf16(false, A, false, B0, (short)0, acc0, false, false);
        acc1 = __builtin_amdgcn_wmma_f32_16x16x32_bf16(false, A, false, B1, (short)0, acc1, false, false);
        acc2 = __builtin_amdgcn_wmma_f32_16x16x32_bf16(false, A, false, B2, (short)0, acc2, false, false);
        acc3 = __builtin_amdgcn_wmma_f32_16x16x32_bf16(false, A, false, B3, (short)0, acc3, false, false);
    }
    // D layout: VGPR r -> M = r + 8*half, N = mn (+ 16*j per sub-tile)
    float* crow = Cout + (size_t)(tm * 16 + half * 8) * 1024 + tn * 64 + mn;
    #pragma unroll
    for (int r = 0; r < 8; r++) {
        crow[(size_t)r * 1024 +  0] = acc0[r];
        crow[(size_t)r * 1024 + 16] = acc1[r];
        crow[(size_t)r * 1024 + 32] = acc2[r];
        crow[(size_t)r * 1024 + 48] = acc3[r];
    }
}

// ---------------- depthwise 3x3 conv + bias + SiLU, -> xc (B, L, DIM) ----------------
__global__ void __launch_bounds__(256)
dwconv_silu(const float* __restrict__ xz, const float* __restrict__ cw,
            const float* __restrict__ cb, float* __restrict__ xc) {
    int idx = blockIdx.x * 256 + threadIdx.x;        // (b*L + l)*DIM + d, d fastest
    if (idx >= NPOS * DIM) return;
    int d  = idx & (DIM - 1);
    int pl = idx >> 9;                               // b*L + l
    int l  = pl % L_;
    int h  = l / W_, w = l % W_;
    const float* wk = cw + (size_t)d * 9;
    float acc = cb[d];
    #pragma unroll
    for (int i = 0; i < 3; i++) {
        int hh = h + i - 1;
        if ((unsigned)hh >= (unsigned)H_) continue;
        #pragma unroll
        for (int j = 0; j < 3; j++) {
            int ww = w + j - 1;
            if ((unsigned)ww >= (unsigned)W_) continue;
            acc += wk[i * 3 + j] * xz[((size_t)(pl + (hh - h) * W_ + (ww - w))) * 1024 + d];
        }
    }
    xc[idx] = silu(acc);
}

// ---------------- x_dbl projection: 512 -> 6 per (b,k,l), wave reductions ----------------
__global__ void __launch_bounds__(256)
xdbl_kernel(const float* __restrict__ xc, const float* __restrict__ xp_w, // (K,6,512) for this expert
            float* __restrict__ dts_raw, float* __restrict__ Bs, float* __restrict__ Cs) {
    int wave = threadIdx.x >> 5, lane = threadIdx.x & 31;
    int id = blockIdx.x * 8 + wave;                  // over B*K*L = 25088
    if (id >= B_ * K_ * L_) return;
    int l = id % L_, bk = id / L_;
    int k = bk % K_, b = bk / K_;
    int lm = scan_map(k, l);
    const float* u   = xc + ((size_t)b * L_ + lm) * DIM;
    const float* wkk = xp_w + (size_t)k * 6 * DIM;
    float a[6] = { 0.f, 0.f, 0.f, 0.f, 0.f, 0.f };
    for (int d = lane; d < DIM; d += 32) {
        float uv = u[d];
        #pragma unroll
        for (int c = 0; c < 6; c++) a[c] += uv * wkk[c * DIM + d];
    }
    #pragma unroll
    for (int c = 0; c < 6; c++)
        #pragma unroll
        for (int m = 16; m >= 1; m >>= 1) a[c] += __shfl_xor(a[c], m, 32);
    if (lane == 0) {
        #pragma unroll
        for (int r = 0; r < R_; r++) dts_raw[((size_t)bk * R_ + r) * L_ + l] = a[r];
        Bs[(size_t)bk * L_ + l] = a[4];
        Cs[(size_t)bk * L_ + l] = a[5];
    }
}

// ---------------- selective scan: one block per (b,k), thread = channel d ----------------
__global__ void __launch_bounds__(512)
ss_scan(const float* __restrict__ xc, const float* __restrict__ dts_raw,
        const float* __restrict__ Bs, const float* __restrict__ Cs,
        const float* __restrict__ dtw, const float* __restrict__ dtb,
        const float* __restrict__ Alog, const float* __restrict__ Dsv,
        float* __restrict__ ysum) {
    __shared__ float s_dt[R_ * L_];
    __shared__ float s_B[L_];
    __shared__ float s_C[L_];
    int bk = blockIdx.x;
    int k = bk % K_, b = bk / K_;
    int d = threadIdx.x;
    for (int i = threadIdx.x; i < R_ * L_; i += blockDim.x)
        s_dt[i] = dts_raw[(size_t)bk * R_ * L_ + i];
    for (int i = threadIdx.x; i < L_; i += blockDim.x) {
        s_B[i] = Bs[(size_t)bk * L_ + i];
        s_C[i] = Cs[(size_t)bk * L_ + i];
    }
    __syncthreads();
    size_t kd = (size_t)k * DIM + d;
    float w0 = dtw[kd * R_ + 0], w1 = dtw[kd * R_ + 1];
    float w2 = dtw[kd * R_ + 2], w3 = dtw[kd * R_ + 3];
    float bb = dtb[kd];
    float A  = -__expf(Alog[kd]);
    float Dd = Dsv[kd];
    float h = 0.f;
    for (int l = 0; l < L_; l++) {
        int lm = scan_map(k, l);
        float dtr = bb + w0 * s_dt[l] + w1 * s_dt[L_ + l] + w2 * s_dt[2 * L_ + l] + w3 * s_dt[3 * L_ + l];
        float dt  = (dtr > 0.f) ? (dtr + log1pf(__expf(-dtr))) : log1pf(__expf(dtr));  // softplus
        float u   = xc[((size_t)b * L_ + lm) * DIM + d];
        h = __expf(dt * A) * h + dt * u * s_B[l];
        float y = h * s_C[l] + u * Dd;
        atomicAdd(&ysum[((size_t)b * L_ + lm) * DIM + d], y);  // merge 4 directions
    }
}

// ---------------- layernorm + SiLU(z) gate + top-2 mix accumulate ----------------
__global__ void __launch_bounds__(256)
norm_gate_mix(const float* __restrict__ ysum, const float* __restrict__ xz,
              const float* __restrict__ nw, const float* __restrict__ nb,
              const float* __restrict__ comb, int e, float* __restrict__ mixed) {
    __shared__ float s1[256];
    __shared__ float s2[256];
    int p = blockIdx.x;
    const float* yr = ysum + (size_t)p * DIM;
    float v0 = yr[threadIdx.x], v1 = yr[threadIdx.x + 256];
    s1[threadIdx.x] = v0 + v1;
    s2[threadIdx.x] = v0 * v0 + v1 * v1;
    __syncthreads();
    for (int s = 128; s > 0; s >>= 1) {
        if (threadIdx.x < s) {
            s1[threadIdx.x] += s1[threadIdx.x + s];
            s2[threadIdx.x] += s2[threadIdx.x + s];
        }
        __syncthreads();
    }
    float mu  = s1[0] / (float)DIM;
    float var = s2[0] / (float)DIM - mu * mu;
    float rs  = rsqrtf(var + 1e-5f);
    float cb  = comb[(size_t)p * E_ + e];
    #pragma unroll
    for (int t = 0; t < 2; t++) {
        int d = threadIdx.x + t * 256;
        float zv  = xz[(size_t)p * 1024 + DIM + d];
        float val = (yr[d] - mu) * rs * nw[d] + nb[d];
        val *= silu(zv);
        mixed[(size_t)p * DIM + d] += val * cb;
    }
}

// ---------------- final: mean over H*W and aux loss ----------------
__global__ void __launch_bounds__(256)
finalize(const float* __restrict__ mixed, const float* __restrict__ aux, float* __restrict__ out) {
    int i = blockIdx.x * 256 + threadIdx.x;
    if (i < B_ * DIM) {
        int b = i / DIM, d = i % DIM;
        float s = 0.f;
        for (int l = 0; l < L_; l++) s += mixed[((size_t)b * L_ + l) * DIM + d];
        out[i] = s / (float)L_;
    }
    if (i == 0) {
        float s = 0.f;
        #pragma unroll
        for (int e = 0; e < E_; e++) {
            float avg  = aux[e] / (float)NPOS;
            float load = aux[E_ + e] / (float)(TOPK_ * NPOS);
            s += avg * load;
        }
        out[B_ * DIM] = AUXW_ * (s / (float)E_) * (float)(E_ * E_);
    }
}

// ---------------- launch ----------------
extern "C" void kernel_launch(void* const* d_in, const int* in_sizes, int n_in,
                              void* d_out, int out_size, void* d_ws, size_t ws_size,
                              hipStream_t stream) {
    const float* x       = (const float*)d_in[0];
    const float* q       = (const float*)d_in[1];
    const float* temp    = (const float*)d_in[2];
    const float* in_proj = (const float*)d_in[3];
    const float* conv_w  = (const float*)d_in[4];
    const float* conv_b  = (const float*)d_in[5];
    const float* x_proj  = (const float*)d_in[6];
    const float* dt_w    = (const float*)d_in[7];
    const float* dt_b    = (const float*)d_in[8];
    const float* A_logs  = (const float*)d_in[9];
    const float* Dsv     = (const float*)d_in[10];
    const float* nw      = (const float*)d_in[11];
    const float* nb      = (const float*)d_in[12];
    float* out = (float*)d_out;

    char* ws = (char*)d_ws;
    size_t off = 0;
    auto alloc = [&](size_t bytes) -> void* {
        void* p = ws + off;
        off = (off + bytes + 255) & ~(size_t)255;
        return p;
    };
    float* comb     = (float*)alloc((size_t)NPOS * E_ * 4);
    float* aux      = (float*)alloc(8 * 4);
    float* xz       = (float*)alloc((size_t)NPOS * 1024 * 4);
    float* xc       = (float*)alloc((size_t)NPOS * DIM * 4);
    float* dts_raw  = (float*)alloc((size_t)B_ * K_ * R_ * L_ * 4);
    float* Bsb      = (float*)alloc((size_t)B_ * K_ * L_ * 4);
    float* Csb      = (float*)alloc((size_t)B_ * K_ * L_ * 4);
    float* ysum     = (float*)alloc((size_t)NPOS * DIM * 4);
    float* mixed    = (float*)alloc((size_t)NPOS * DIM * 4);
    unsigned short* xb = (unsigned short*)alloc((size_t)NPOS * DIM * 2);
    unsigned short* wb = (unsigned short*)alloc((size_t)E_ * 1024 * DIM * 2);

    const int nElem = NPOS * DIM;                    // 3.21M
    zero_f32<<<(nElem + 255) / 256, 256, 0, stream>>>(mixed, nElem);
    zero_f32<<<1, 8, 0, stream>>>(aux, 8);
    gating_kernel<<<(NPOS + 7) / 8, 256, 0, stream>>>(x, q, temp, comb, aux);
    cast_bf16<<<(nElem + 255) / 256, 256, 0, stream>>>(x, xb, nElem);
    cast_bf16<<<(E_ * 1024 * DIM + 255) / 256, 256, 0, stream>>>(in_proj, wb, E_ * 1024 * DIM);

    for (int e = 0; e < E_; e++) {
        gemm_xz_bf16<<<(392 * 16) / 4, 128, 0, stream>>>(xb, wb + (size_t)e * 1024 * DIM, xz);
        dwconv_silu<<<(nElem + 255) / 256, 256, 0, stream>>>(
            xz, conv_w + (size_t)e * DIM * 9, conv_b + (size_t)e * DIM, xc);
        zero_f32<<<(nElem + 255) / 256, 256, 0, stream>>>(ysum, nElem);
        xdbl_kernel<<<(B_ * K_ * L_ + 7) / 8, 256, 0, stream>>>(
            xc, x_proj + (size_t)e * K_ * 6 * DIM, dts_raw, Bsb, Csb);
        ss_scan<<<B_ * K_, 512, 0, stream>>>(
            xc, dts_raw, Bsb, Csb,
            dt_w + (size_t)e * K_ * DIM * R_, dt_b + (size_t)e * K_ * DIM,
            A_logs + (size_t)e * K_ * DIM, Dsv + (size_t)e * K_ * DIM, ysum);
        norm_gate_mix<<<NPOS, 256, 0, stream>>>(
            ysum, xz, nw + (size_t)e * DIM, nb + (size_t)e * DIM, comb, e, mixed);
    }
    finalize<<<(B_ * DIM + 255) / 256, 256, 0, stream>>>(mixed, aux, out);
}
